// MultiHeadCrossAttention_20083267076886
// MI455X (gfx1250) — matrix operand order repaired and checked by reading, not compiled, not run on hardware
//
#include <hip/hip_runtime.h>
#include <hip/hip_bf16.h>

// ---------------------------------------------------------------------------
// Multi-head cross attention on gfx1250 (MI455X), bf16 WMMA pipeline.
//   q = x1 @ Wq^T + bq ; k = x2 @ Wk^T + bk ; v = x2 @ Wv^T + bv
//   attn = softmax(q k^T / sqrt(D)) v ; out = attn @ Wo^T + bo
// All GEMMs run on v_wmma_f32_16x16x32_bf16 (f32 accumulate), softmax in f32.
// Row reductions use DPP16 butterflies (VALU) instead of ds_bpermute.
// ---------------------------------------------------------------------------

typedef __bf16 bf16_t;
typedef __attribute__((ext_vector_type(16))) __bf16 v16bf;
typedef __attribute__((ext_vector_type(8)))  __bf16 v8bf;
typedef __attribute__((ext_vector_type(4)))  __bf16 v4bf;
typedef __attribute__((ext_vector_type(8)))  float  v8f;

#define EMBED 1024
#define NHEAD 16
#define HDIM  64

static __device__ __forceinline__ v8f vzero8() {
    v8f z = {0.f, 0.f, 0.f, 0.f, 0.f, 0.f, 0.f, 0.f};
    return z;
}

// A-matrix (16x32 bf16) fragment load.  Per documented layout, lane L
// (M = L%16) holds K = {base..base+7} and {base+16..base+23} where
// base = (L>=16) ? 8 : 0.  `p` must already point at row-start + k0 + base.
static __device__ __forceinline__ v16bf load_a16(const bf16_t* p) {
    v8bf lo = *(const v8bf*)(p);
    v8bf hi = *(const v8bf*)(p + 16);
    return __builtin_shufflevector(lo, hi, 0, 1, 2, 3, 4, 5, 6, 7,
                                           8, 9, 10, 11, 12, 13, 14, 15);
}

// B-matrix (32x16 bf16) fragment load.  Lane L (N = L%16) holds
// K = {base..base+7} and {base+16..base+23}, base = half*8.  `p` points at col-row + k0 + half*8.
static __device__ __forceinline__ v16bf load_b16(const bf16_t* p) {
    v8bf lo = *(const v8bf*)(p);
    v8bf hi = *(const v8bf*)(p + 16);
    return __builtin_shufflevector(lo, hi, 0, 1, 2, 3, 4, 5, 6, 7,
                                           8, 9, 10, 11, 12, 13, 14, 15);
}

static __device__ __forceinline__ v8f wmma_bf16(v16bf a, v16bf b, v8f c) {
    return __builtin_amdgcn_wmma_f32_16x16x32_bf16(
        /*neg_a=*/false, a, /*neg_b=*/false, b,
        /*c_mod=*/(short)0, c, /*reuse_a=*/false, /*reuse_b=*/false);
}

// ---------------------------------------------------------------------------
// DPP16 butterfly reductions across 16-lane groups (pure VALU, no LDS).
//   0xB1  = quad_perm [1,0,3,2]  (xor 1)
//   0x4E  = quad_perm [2,3,0,1]  (xor 2)
//   0x141 = row_half_mirror      (combines 4-groups once they are uniform)
//   0x140 = row_mirror           (combines 8-groups once they are uniform)
// ---------------------------------------------------------------------------
template <int CTRL>
static __device__ __forceinline__ float dpp_mov_f32(float v) {
    return __int_as_float(__builtin_amdgcn_update_dpp(
        0, __float_as_int(v), CTRL, 0xF, 0xF, true));
}

static __device__ __forceinline__ float rowmax16(float v) {
    v = fmaxf(v, dpp_mov_f32<0xB1>(v));
    v = fmaxf(v, dpp_mov_f32<0x4E>(v));
    v = fmaxf(v, dpp_mov_f32<0x141>(v));
    v = fmaxf(v, dpp_mov_f32<0x140>(v));
    return v;
}

static __device__ __forceinline__ float rowsum16(float v) {
    v += dpp_mov_f32<0xB1>(v);
    v += dpp_mov_f32<0x4E>(v);
    v += dpp_mov_f32<0x141>(v);
    v += dpp_mov_f32<0x140>(v);
    return v;
}

// ---------------------------------------------------------------------------
// f32 -> bf16 conversion (vectorized x4)
// ---------------------------------------------------------------------------
__global__ __launch_bounds__(256)
void cvt_f32_bf16_kernel(const float* __restrict__ in, bf16_t* __restrict__ out, int n4) {
    int i = blockIdx.x * blockDim.x + threadIdx.x;
    if (i < n4) {
        float4 f = ((const float4*)in)[i];
        v4bf o;
        o[0] = (bf16_t)f.x; o[1] = (bf16_t)f.y;
        o[2] = (bf16_t)f.z; o[3] = (bf16_t)f.w;
        ((v4bf*)out)[i] = o;
    }
}

// ---------------------------------------------------------------------------
// GEMM: Y[M,N] = X[M,K=1024] * W[N,K=1024]^T + bias[N]
// Block = 128 threads = 4 waves; each wave computes a 64x64 tile (4x4 WMMA
// accumulator fragments).  Grid: (N/128, M/128).
// MODE 0: store bf16 to [B,H,S,D]       (Q, K)
// MODE 2: store bf16 to [B,H,D,S]       (V transposed for PV GEMM)
// MODE 1: store f32  to row-major [M,N] (final output)
// S = 1<<LOG2S so batch/seq split is shifts, not runtime division.
// ---------------------------------------------------------------------------
template <int MODE, int LOG2S>
__global__ __launch_bounds__(128)
void proj_kernel(const bf16_t* __restrict__ X, const bf16_t* __restrict__ W,
                 const float* __restrict__ bias, void* __restrict__ out) {
    const int lane = threadIdx.x & 31;
    const int wave = threadIdx.x >> 5;
    const int l16  = lane & 15;
    const int half = lane >> 4;

    const int m0 = blockIdx.y * 128 + (wave >> 1) * 64;
    const int n0 = blockIdx.x * 128 + (wave & 1) * 64;

    v8f c[4][4];
#pragma unroll
    for (int mt = 0; mt < 4; ++mt)
#pragma unroll
        for (int nt = 0; nt < 4; ++nt) c[mt][nt] = vzero8();

    const bf16_t* xrow[4];
    const bf16_t* wrow[4];
#pragma unroll
    for (int t = 0; t < 4; ++t) {
        xrow[t] = X + (size_t)(m0 + t * 16 + l16) * EMBED + half * 8;
        wrow[t] = W + (size_t)(n0 + t * 16 + l16) * EMBED + half * 8;
    }

    for (int k0 = 0; k0 < EMBED; k0 += 32) {
        v16bf a[4], b[4];
#pragma unroll
        for (int mt = 0; mt < 4; ++mt) a[mt] = load_a16(xrow[mt] + k0);
#pragma unroll
        for (int nt = 0; nt < 4; ++nt) b[nt] = load_b16(wrow[nt] + k0);
#pragma unroll
        for (int mt = 0; mt < 4; ++mt)
#pragma unroll
            for (int nt = 0; nt < 4; ++nt)
                c[mt][nt] = wmma_bf16(a[mt], b[nt], c[mt][nt]);
    }

    // Epilogue. C-layout: element r of the v8f holds row (r + 8*half),
    // column l16 within each 16x16 tile.
#pragma unroll
    for (int nt = 0; nt < 4; ++nt) {
        const int n  = n0 + nt * 16 + l16;
        const float bn = bias[n];
#pragma unroll
        for (int mt = 0; mt < 4; ++mt) {
#pragma unroll
            for (int r = 0; r < 8; ++r) {
                const int m = m0 + mt * 16 + r + 8 * half;
                const float y = c[mt][nt][r] + bn;
                if (MODE == 1) {
                    ((float*)out)[(size_t)m * EMBED + n] = y;
                } else {
                    const int bb = m >> LOG2S;             // batch
                    const int s  = m & ((1 << LOG2S) - 1); // seq position
                    const int h  = n >> 6, d = n & 63;
                    size_t idx;
                    if (MODE == 0)
                        idx = (((size_t)(bb * NHEAD + h) << LOG2S) + s) * HDIM + d;
                    else // MODE 2: transposed within head
                        idx = (((size_t)(bb * NHEAD + h) * HDIM + d) << LOG2S) + s;
                    ((bf16_t*)out)[idx] = (bf16_t)y;
                }
            }
        }
    }
}

// ---------------------------------------------------------------------------
// Flash attention. Grid: (Sq/128, B*H); block = 128 threads = 4 waves.
// Each wave owns 32 query rows, streams KV in 64-row tiles.
// Q : [B,H,Sq,64] bf16     K : [B,H,Skv,64] bf16    Vt : [B,H,64,Skv] bf16
// attn out : [B,Sq,E] bf16
// ---------------------------------------------------------------------------
__global__ __launch_bounds__(128)
void attn_kernel(const bf16_t* __restrict__ Q, const bf16_t* __restrict__ K,
                 const bf16_t* __restrict__ Vt, bf16_t* __restrict__ attn,
                 int Sq, int Skv) {
    __shared__ bf16_t lds_p[4][32 * 64];   // per-wave P staging (C->A relayout)

    const int lane = threadIdx.x & 31;
    const int wave = threadIdx.x >> 5;
    const int l16  = lane & 15;
    const int half = lane >> 4;

    const int bh = blockIdx.y;          // b*H + h
    const int b  = bh >> 4;
    const int h  = bh & 15;
    const int q0 = blockIdx.x * 128 + wave * 32;

    const bf16_t* Qbh = Q  + (size_t)bh * Sq  * HDIM;
    const bf16_t* Kbh = K  + (size_t)bh * Skv * HDIM;
    const bf16_t* Vbh = Vt + (size_t)bh * HDIM * Skv;

    // Resident Q fragments: [mt 0..1][kd 0..1] covering 32 rows x 64 (D).
    v16bf qa[2][2];
#pragma unroll
    for (int mt = 0; mt < 2; ++mt)
#pragma unroll
        for (int kd = 0; kd < 2; ++kd)
            qa[mt][kd] = load_a16(Qbh + (size_t)(q0 + mt * 16 + l16) * HDIM +
                                  kd * 32 + half * 8);

    v8f o[2][4];
#pragma unroll
    for (int mt = 0; mt < 2; ++mt)
#pragma unroll
        for (int dt = 0; dt < 4; ++dt) o[mt][dt] = vzero8();

    float mrow[2][8], lrow[2][8];
#pragma unroll
    for (int mt = 0; mt < 2; ++mt)
#pragma unroll
        for (int r = 0; r < 8; ++r) { mrow[mt][r] = -1.0e30f; lrow[mt][r] = 0.f; }

    const float scale = 0.125f;  // 1/sqrt(64)

    for (int kv0 = 0; kv0 < Skv; kv0 += 64) {
        if (kv0 + 64 < Skv)  // emits global_prefetch_b8 for next K tile
            __builtin_prefetch(Kbh + (size_t)(kv0 + 64 + l16) * HDIM, 0, 1);

        // ---- S = Q K^T for a 32x64 score tile ----
        v8f s[2][4];
#pragma unroll
        for (int nt = 0; nt < 4; ++nt) {
            const bf16_t* kr = Kbh + (size_t)(kv0 + nt * 16 + l16) * HDIM + half * 8;
            v16bf kb0 = load_b16(kr);
            v16bf kb1 = load_b16(kr + 32);
#pragma unroll
            for (int mt = 0; mt < 2; ++mt) {
                v8f z = vzero8();
                z = wmma_bf16(qa[mt][0], kb0, z);
                z = wmma_bf16(qa[mt][1], kb1, z);
                s[mt][nt] = z;
            }
        }

        // ---- online softmax (row groups live on 16-lane halves) ----
#pragma unroll
        for (int mt = 0; mt < 2; ++mt) {
#pragma unroll
            for (int r = 0; r < 8; ++r) {
                float v = fmaxf(fmaxf(s[mt][0][r], s[mt][1][r]),
                                fmaxf(s[mt][2][r], s[mt][3][r])) * scale;
                v = rowmax16(v);                       // DPP16 butterfly (VALU)
                const float mnew  = fmaxf(mrow[mt][r], v);
                const float alpha = __expf(mrow[mt][r] - mnew);
                mrow[mt][r] = mnew;
                lrow[mt][r] *= alpha;
#pragma unroll
                for (int dt = 0; dt < 4; ++dt) o[mt][dt][r] *= alpha;

                float psum = 0.f;
#pragma unroll
                for (int nt = 0; nt < 4; ++nt) {
                    const float p = __expf(s[mt][nt][r] * scale - mnew);
                    psum += p;
                    lds_p[wave][(mt * 16 + r + 8 * half) * 64 + nt * 16 + l16] =
                        (bf16_t)p;
                }
                lrow[mt][r] += rowsum16(psum);         // DPP16 butterfly (VALU)
            }
        }
        __syncthreads();   // LDS P visible in A layout

        // ---- O += P V ----
#pragma unroll
        for (int kp = 0; kp < 2; ++kp) {
            v16bf pa0 = load_a16(&lds_p[wave][(l16) * 64 + kp * 32 + half * 8]);
            v16bf pa1 = load_a16(&lds_p[wave][(16 + l16) * 64 + kp * 32 + half * 8]);
#pragma unroll
            for (int dt = 0; dt < 4; ++dt) {
                v16bf vb = load_b16(Vbh + (size_t)(dt * 16 + l16) * Skv +
                                    kv0 + kp * 32 + half * 8);
                o[0][dt] = wmma_bf16(pa0, vb, o[0][dt]);
                o[1][dt] = wmma_bf16(pa1, vb, o[1][dt]);
            }
        }
        __syncthreads();
    }

    // ---- normalize and scatter to [B, Sq, E] bf16 ----
#pragma unroll
    for (int mt = 0; mt < 2; ++mt) {
#pragma unroll
        for (int r = 0; r < 8; ++r) {
            const float inv  = 1.0f / lrow[mt][r];
            const int   qrow = q0 + mt * 16 + r + 8 * half;
#pragma unroll
            for (int dt = 0; dt < 4; ++dt) {
                const int col = h * 64 + dt * 16 + l16;
                attn[((size_t)b * Sq + qrow) * EMBED + col] =
                    (bf16_t)(o[mt][dt][r] * inv);
            }
        }
    }
}

// ---------------------------------------------------------------------------
// Host-side launcher
// ---------------------------------------------------------------------------
extern "C" void kernel_launch(void* const* d_in, const int* in_sizes, int n_in,
                              void* d_out, int out_size, void* d_ws, size_t ws_size,
                              hipStream_t stream) {
    (void)in_sizes; (void)n_in; (void)out_size; (void)ws_size;

    const float* x1 = (const float*)d_in[0];
    const float* x2 = (const float*)d_in[1];
    const float* Wq = (const float*)d_in[2];
    const float* bq = (const float*)d_in[3];
    const float* Wk = (const float*)d_in[4];
    const float* bk = (const float*)d_in[5];
    const float* Wv = (const float*)d_in[6];
    const float* bv = (const float*)d_in[7];
    const float* Wo = (const float*)d_in[8];
    const float* bo = (const float*)d_in[9];

    const int B = 4, Sq = 1024, Skv = 2048;
    const size_t nx1 = (size_t)B * Sq * EMBED;    // 4M
    const size_t nx2 = (size_t)B * Skv * EMBED;   // 8M
    const size_t nw  = (size_t)EMBED * EMBED;     // 1M

    size_t off = 0;
    auto carve = [&](size_t elems) {
        bf16_t* p = (bf16_t*)((char*)d_ws + off);
        off += ((elems * sizeof(bf16_t)) + 255) & ~(size_t)255;
        return p;
    };
    bf16_t* x1b  = carve(nx1);
    bf16_t* x2b  = carve(nx2);
    bf16_t* wqb  = carve(nw);
    bf16_t* wkb  = carve(nw);
    bf16_t* wvb  = carve(nw);
    bf16_t* wob  = carve(nw);
    bf16_t* Qb   = carve(nx1);   // [B,H,Sq,D]
    bf16_t* Kb   = carve(nx2);   // [B,H,Skv,D]
    bf16_t* Vtb  = carve(nx2);   // [B,H,D,Skv]
    bf16_t* attb = carve(nx1);   // [B,Sq,E]

    // 1) f32 -> bf16 conversions
    auto cvt = [&](const float* src, bf16_t* dst, size_t n) {
        int n4 = (int)(n / 4);
        cvt_f32_bf16_kernel<<<(n4 + 255) / 256, 256, 0, stream>>>(src, dst, n4);
    };
    cvt(x1, x1b, nx1);
    cvt(x2, x2b, nx2);
    cvt(Wq, wqb, nw);
    cvt(Wk, wkb, nw);
    cvt(Wv, wvb, nw);
    cvt(Wo, wob, nw);

    // 2) Q/K/V projections (WMMA GEMMs)
    {
        dim3 blk(128);
        dim3 gq(EMBED / 128, (B * Sq) / 128);
        dim3 gk(EMBED / 128, (B * Skv) / 128);
        proj_kernel<0, 10><<<gq, blk, 0, stream>>>(x1b, wqb, bq, Qb);   // Q
        proj_kernel<0, 11><<<gk, blk, 0, stream>>>(x2b, wkb, bk, Kb);   // K
        proj_kernel<2, 11><<<gk, blk, 0, stream>>>(x2b, wvb, bv, Vtb);  // V^T
    }

    // 3) flash attention
    {
        dim3 blk(128);
        dim3 ga(Sq / 128, B * NHEAD);
        attn_kernel<<<ga, blk, 0, stream>>>(Qb, Kb, Vtb, attb, Sq, Skv);
    }

    // 4) output projection -> f32 d_out
    {
        dim3 blk(128);
        dim3 go(EMBED / 128, (B * Sq) / 128);
        proj_kernel<1, 10><<<go, blk, 0, stream>>>(attb, wob, bo, d_out);
    }
}
